// MultiSpeciesHead_83322365542842
// MI455X (gfx1250) — matrix operand order repaired and checked by reading, not compiled, not run on hardware
//
#include <hip/hip_runtime.h>

// ---- problem constants (match reference) ----
constexpr int NB = 8;     // batch
constexpr int NS = 2048;  // sequence (M total per batch)
constexpr int ND = 1536;  // model dim (K)
constexpr int NT = 2048;  // padded tracks (N)

constexpr int BM = 256;   // block tile M
constexpr int BN = 128;   // block tile N
constexpr int KBLK = 32;  // K per bf16 WMMA stage
constexpr int LDSW = 40;  // LDS row stride in halfs (32 + 8 pad, 80B, 16B aligned)
constexpr int KIT = ND / KBLK;  // 48

typedef __bf16 bf16_t;
typedef __attribute__((ext_vector_type(16))) __bf16 v16bf;
typedef __attribute__((ext_vector_type(8)))  __bf16 v8bf;
typedef __attribute__((ext_vector_type(4)))  __bf16 v4bf;
typedef __attribute__((ext_vector_type(8)))  float  v8f;
typedef __attribute__((ext_vector_type(4)))  float  v4f;

__global__ __launch_bounds__(256)
void msh_gemm_bf16_kernel(const float* __restrict__ x,
                          const int*   __restrict__ head_idx,
                          const float* __restrict__ W,
                          const float* __restrict__ bias,
                          float*       __restrict__ out) {
    // double-buffered staging tiles (bf16): A 2x20KB, B 2x10KB = 60KB
    __shared__ __align__(16) bf16_t Als[2][BM * LDSW];  // [row 0..255][k 0..31]
    __shared__ __align__(16) bf16_t Bls[2][BN * LDSW];  // transposed: [t 0..127][k 0..31]

    const int tid    = threadIdx.x;
    const int lane   = tid & 31;
    const int wid    = tid >> 5;
    const int wave_m = wid & 3;   // 4 waves along M: 64 rows each
    const int wave_n = wid >> 2;  // 2 waves along N: 64 cols each
    const int lgrp   = lane >> 4; // 0 = lanes 0-15, 1 = lanes 16-31
    const int lmod   = lane & 15;

    const int n0 = blockIdx.x * BN;
    const int m0 = blockIdx.y * BM;
    const int bz = blockIdx.z;

    const int h = head_idx[bz];
    const float* __restrict__ xb = x    + (size_t)bz * NS * ND;
    const float* __restrict__ Wh = W    + (size_t)h  * ND * NT;
    const float* __restrict__ bh = bias + (size_t)h  * NT;

    // ---- global staging assignments ----
    // A: 256 rows x 32 k floats; each thread: 8 x float4 (rows tid/8 + 32*i)
    const int a_row  = tid >> 3;
    const int a_col4 = (tid & 7) * 4;
    // B: 32 k-rows x 128 t floats; each thread owns a 4k x 4t micro-block
    const int b_kb = (tid >> 5) * 4;  // k base 0..28
    const int b_tb = (tid & 31) * 4;  // t base 0..124

    // Running base pointers: all 12 loads per step use compile-time immediate
    // offsets from these two bases (A row stride 32*ND*4 = 196608B < 2^23, so
    // the 8 A loads fold into IOFFSET; pointers advance by one K-step each call).
    const float* aptr = xb + (size_t)(m0 + a_row) * ND + a_col4;
    const float* bptr = Wh + (size_t)b_kb * NT + n0 + b_tb;

    v4f ar[8], br[4];

    auto gload = [&]() {
#pragma unroll
        for (int i = 0; i < 8; ++i)
            ar[i] = *(const v4f*)(aptr + (size_t)i * 32 * ND);
#pragma unroll
        for (int r = 0; r < 4; ++r)
            br[r] = *(const v4f*)(bptr + (size_t)r * NT);
        aptr += KBLK;               // 128 bytes
        bptr += (size_t)KBLK * NT;  // 32 W rows
    };

    auto stage = [&](int buf) {
#pragma unroll
        for (int i = 0; i < 8; ++i) {
            v4bf p;
            p.x = (bf16_t)ar[i].x; p.y = (bf16_t)ar[i].y;
            p.z = (bf16_t)ar[i].z; p.w = (bf16_t)ar[i].w;
            *(v4bf*)(&Als[buf][(a_row + i * 32) * LDSW + a_col4]) = p;  // ds_store_b64
        }
#pragma unroll
        for (int j = 0; j < 4; ++j) {  // 4x4 register micro-transpose of W tile
            v4bf p;
            p.x = (bf16_t)br[0][j]; p.y = (bf16_t)br[1][j];
            p.z = (bf16_t)br[2][j]; p.w = (bf16_t)br[3][j];
            *(v4bf*)(&Bls[buf][(b_tb + j) * LDSW + b_kb]) = p;          // ds_store_b64
        }
    };

    // ---- accumulators initialized with the per-column bias ----
    // C/D layout: each lane owns column N = lane%16; bias constant over its 8 rows.
    v8f acc[4][4];
#pragma unroll
    for (int ni = 0; ni < 4; ++ni) {
        const float bv = bh[n0 + wave_n * 64 + ni * 16 + lmod];
        const v8f bsplat = {bv, bv, bv, bv, bv, bv, bv, bv};
#pragma unroll
        for (int mi = 0; mi < 4; ++mi) acc[mi][ni] = bsplat;
    }

    // ---- pipeline prolog ----
    gload();
    stage(0);
    gload();
    __syncthreads();

    // ---- main K loop: one barrier per step, double-buffered LDS ----
    for (int kt = 0; kt < KIT; ++kt) {
        const int cur = kt & 1;

        // A fragments (documented 16-bit A 16x32 layout):
        // lane<16 : halfs[0..7]=K0..7  @ +0B,  halfs[8..15]=K16..23 @ +32B
        // lane>=16: halfs[0..7]=K8..15 @ +16B, halfs[8..15]=K24..31 @ +48B
        v16bf af[4];
#pragma unroll
        for (int mi = 0; mi < 4; ++mi) {
            const bf16_t* rowp = &Als[cur][(wave_m * 64 + mi * 16 + lmod) * LDSW];
            v8bf lo = *(const v8bf*)(rowp + (lgrp ? 8 : 0));
            v8bf hi = *(const v8bf*)(rowp + (lgrp ? 24 : 16));
            af[mi] = __builtin_shufflevector(lo, hi, 0, 1, 2, 3, 4, 5, 6, 7,
                                             8, 9, 10, 11, 12, 13, 14, 15);
        }

        // B fragments: lanes 0-15 hold K0..7,K16..23 of column lane%16, lanes 16-31 K8..15,K24..31
#pragma unroll
        for (int ni = 0; ni < 4; ++ni) {
            const bf16_t* rowp =
                &Bls[cur][(wave_n * 64 + ni * 16 + lmod) * LDSW];
            v8bf lo = *(const v8bf*)(rowp + (lgrp ? 8 : 0));
            v8bf hi = *(const v8bf*)(rowp + (lgrp ? 24 : 16));
            v16bf bf = __builtin_shufflevector(lo, hi, 0, 1, 2, 3, 4, 5, 6, 7,
                                               8, 9, 10, 11, 12, 13, 14, 15);
#pragma unroll
            for (int mi = 0; mi < 4; ++mi) {
                acc[mi][ni] = __builtin_amdgcn_wmma_f32_16x16x32_bf16(
                    false, af[mi], false, bf,
                    (short)0, acc[mi][ni], false, false);
            }
        }

        // convert + store next stage into the other buffer, prefetch the one after
        if (kt + 1 < KIT) stage(1 - cur);
        if (kt + 2 < KIT) gload();
        __syncthreads();
    }

    // ---- epilogue: non-temporal stores (bias already folded in) ----
    // Output (134MB) is write-once; keep it from evicting x/W tiles in L2.
    const int madd = lgrp * 8;
#pragma unroll
    for (int mi = 0; mi < 4; ++mi) {
#pragma unroll
        for (int ni = 0; ni < 4; ++ni) {
            const int t = n0 + wave_n * 64 + ni * 16 + lmod;
            const int s_base = m0 + wave_m * 64 + mi * 16 + madd;
            float* op = out + ((size_t)bz * NS + s_base) * NT + t;
#pragma unroll
            for (int r = 0; r < 8; ++r)
                __builtin_nontemporal_store(acc[mi][ni][r], op + (size_t)r * NT);
        }
    }
}

extern "C" void kernel_launch(void* const* d_in, const int* in_sizes, int n_in,
                              void* d_out, int out_size, void* d_ws, size_t ws_size,
                              hipStream_t stream) {
    const float* x        = (const float*)d_in[0];
    const int*   head_idx = (const int*)  d_in[1];
    const float* W        = (const float*)d_in[2];
    const float* bias     = (const float*)d_in[3];
    float*       out      = (float*)d_out;

    dim3 grid(NT / BN, NS / BM, NB);  // (16, 8, 8)
    msh_gemm_bf16_kernel<<<grid, 256, 0, stream>>>(x, head_idx, W, bias, out);
}
